// SynoBlock_25958782337899
// MI455X (gfx1250) — matrix operand order repaired and checked by reading, not compiled, not run on hardware
//
#include <hip/hip_runtime.h>
#include <hip/hip_bf16.h>
#include <math.h>

typedef __attribute__((ext_vector_type(16))) _Float16 v16h;
typedef __attribute__((ext_vector_type(8)))  _Float16 h8;
typedef __attribute__((ext_vector_type(8)))  float    v8f;

#define NB   8
#define NT   16
#define NP   14
#define NH   16
#define ND   64
#define NL   196
#define NDM  1024
#define NNS  16
#define NT2  256

__device__ __forceinline__ v16h mk16(h8 lo, h8 hi) {
  v16h r;
#pragma unroll
  for (int i = 0; i < 8; ++i) { r[i] = lo[i]; r[8 + i] = hi[i]; }
  return r;
}

__device__ __forceinline__ h8 ldg_h8(const float* p) {
  h8 r;
#pragma unroll
  for (int i = 0; i < 8; ++i) r[i] = (_Float16)p[i];
  return r;
}

__device__ __forceinline__ v8f wmma16(v16h a, v16h b, v8f c) {
  // D = A(16x32,f16) x B(32x16,f16) + C(16x16,f32)
  return __builtin_amdgcn_wmma_f32_16x16x32_f16(false, a, false, b, (short)0, c,
                                                false, false);
}

// ---------------------------------------------------------------------------
// Kernel 1: temporal affinity. One block per (b,l). 32 grams (16 q-heads +
// 16 k-heads); each gram = 16x16x64 GEMM via 2 chained WMMAs, key-axis
// softmax (mask-major batched butterflies + v_rcp normalization), and the
// 3x3/32ch temporal conv evaluated IN REGISTERS on the WMMA C-layout
// (lane = key col, VGPR = query row): column taps = 1 shuffle/reg, row taps
// = adjacent VGPRs, the row-7/8 seam = one xor-16 exchange. Per-wave
// partials reduced via 8KB LDS. Writes aff0[b*L+l][256] == [B,P,P,T2] flat.
// ---------------------------------------------------------------------------
__global__ __launch_bounds__(256) void k_temporal(
    const float* __restrict__ q, const float* __restrict__ kk,
    const float* __restrict__ tw, const float* __restrict__ tb,
    float* __restrict__ aff0) {
  __shared__ _Float16 slab[8][16 * 64];   // per-wave gram staging (f16), 16 KB
  __shared__ float    part[8][256];       // per-wave conv partials, 8 KB

  const int tid  = threadIdx.x;
  const int wave = tid >> 5;
  const int lane = tid & 31;
  const int bl   = blockIdx.x;
  const int b    = bl / NL;
  const int l    = bl - b * NL;

  const int half = lane >> 4;
  const int m    = lane & 15;              // key col / matrix row id
  const int grp  = lane & 16;              // group base for 16-lane shuffles
  const int slx  = grp | ((m - 1) & 15);   // src lane for col n-1
  const int srx  = grp | ((m + 1) & 15);   // src lane for col n+1

  float o[8];
#pragma unroll
  for (int j = 0; j < 8; ++j) o[j] = 0.f;

  for (int iter = 0; iter < 4; ++iter) {
    const int c = wave * 4 + iter;         // channel 0..31 (uniform per wave)
    const int h = c & 15;
    const float* src  = (c < 16) ? q : kk;
    const float* base = src + ((size_t)(b * NT) * 197 + (l + 1)) * (size_t)NDM
                            + (size_t)h * ND;    // [t][d], t-stride 197*1024
    // stage 16x64 f32 -> f16 LDS: lane loads row lane>>1, cols (lane&1)*32..+31
    {
      const int r  = lane >> 1;
      const int c0 = (lane & 1) * 32;
      const float* rp = base + (size_t)r * 197 * NDM + c0;
      _Float16* dp = &slab[wave][r * 64 + c0];
#pragma unroll
      for (int i = 0; i < 32; ++i) dp[i] = (_Float16)rp[i];
    }
    __builtin_amdgcn_wave_barrier();       // same-wave DS ordering is in-order

    const _Float16* row = &slab[wave][m * 64];   // this lane's matrix row
    v8f acc = {};
#pragma unroll
    for (int k0 = 0; k0 < 64; k0 += 32) {
      // A operand: lane half picks K 0-15 vs 16-31
      h8 x0 = *(const h8*)(row + k0 + half * 16);
      h8 x1 = *(const h8*)(row + k0 + half * 16 + 8);
      // B operand = A^T: lane=N holds its row's K 0-15 / 16-31
      h8 y0 = *(const h8*)(row + k0 + half * 16);
      h8 y1 = *(const h8*)(row + k0 + half * 16 + 8);
      acc = wmma16(mk16(x0, x1), mk16(y0, y1), acc);
    }
    __builtin_amdgcn_wave_barrier();       // keep next-iter stores after reads

    // softmax over key frames (lanes within 16-group); logits ~N(0,1) so no
    // max-shift required. Mask-major butterflies: 8 independent ds_bpermutes
    // per step so the LDS-crossbar latency overlaps; v_rcp instead of a full
    // IEEE divide (softmax tolerance).
    float e[8], s[8];
#pragma unroll
    for (int j = 0; j < 8; ++j) { e[j] = __expf(acc[j] * 0.125f); s[j] = e[j]; }
#pragma unroll
    for (int msk = 1; msk <= 8; msk <<= 1) {
      float t0[8];
#pragma unroll
      for (int j = 0; j < 8; ++j) t0[j] = __shfl_xor(s[j], msk);
#pragma unroll
      for (int j = 0; j < 8; ++j) s[j] += t0[j];
    }
    float p[8];
#pragma unroll
    for (int j = 0; j < 8; ++j) p[j] = e[j] * __builtin_amdgcn_rcpf(s[j]);

    // 3x3 conv tap accumulation in registers (zero-padded 16x16 plane).
    float w9[9];
#pragma unroll
    for (int i = 0; i < 9; ++i) w9[i] = tw[c * 9 + i];

    float pl[8], pr[8];
#pragma unroll
    for (int j = 0; j < 8; ++j) {
      const float xl = __shfl(p[j], slx);
      const float xr = __shfl(p[j], srx);
      pl[j] = (m == 0)  ? 0.f : xl;        // value at col n-1
      pr[j] = (m == 15) ? 0.f : xr;        // value at col n+1
    }
    // seam row exchange: half0 sends row7 (p[7]) / receives row8;
    //                    half1 sends row8 (p[0]) / receives row7.
    const float ec = __shfl_xor(half ? p[0] : p[7], 16);
    float el = __shfl(ec, slx);
    float er = __shfl(ec, srx);
    el = (m == 0)  ? 0.f : el;
    er = (m == 15) ? 0.f : er;

#pragma unroll
    for (int j = 0; j < 8; ++j) {
      float a2 = w9[3] * pl[j] + w9[4] * p[j] + w9[5] * pr[j];      // dy=0
      if (j >= 1)                                                    // dy=-1
        a2 += w9[0] * pl[j - 1] + w9[1] * p[j - 1] + w9[2] * pr[j - 1];
      else if (half)                       // row 8's top neighbor = row 7
        a2 += w9[0] * el + w9[1] * ec + w9[2] * er;
      if (j <= 6)                                                    // dy=+1
        a2 += w9[6] * pl[j + 1] + w9[7] * p[j + 1] + w9[8] * pr[j + 1];
      else if (!half)                      // row 7's bottom neighbor = row 8
        a2 += w9[6] * el + w9[7] * ec + w9[8] * er;
      o[j] += a2;
    }
  }

  // cross-wave reduction of the 8 partial output planes
#pragma unroll
  for (int j = 0; j < 8; ++j) part[wave][(j + half * 8) * 16 + m] = o[j];
  __syncthreads();
  float r = tb[0];
#pragma unroll
  for (int w = 0; w < 8; ++w) r += part[w][tid];
  aff0[(size_t)bl * 256 + tid] = r;
}

// ---------------------------------------------------------------------------
// Kernel 2: LayerNorm + MLP(256->256->256, exact GELU) + residual.
// 16 rows per block; two WMMA GEMMs (f16 in, f32 acc), weights cast on load.
// ---------------------------------------------------------------------------
__global__ __launch_bounds__(256) void k_mlp(
    const float* __restrict__ aff0, const float* __restrict__ lng,
    const float* __restrict__ lnb, const float* __restrict__ w1,
    const float* __restrict__ b1, const float* __restrict__ w2,
    const float* __restrict__ b2, float* __restrict__ aff1) {
  __shared__ float    xrow[16][256];    // original rows (residual), 16 KB
  __shared__ _Float16 hbuf[16][256];    // LN output, 8 KB
  __shared__ _Float16 gbuf[16][256];    // GELU output, 8 KB

  const int tid  = threadIdx.x;
  const int wave = tid >> 5;
  const int lane = tid & 31;
  const int half = lane >> 4;
  const int m    = lane & 15;
  const size_t row0 = (size_t)blockIdx.x * 16;

  for (int i = tid; i < 4096; i += 256) xrow[i >> 8][i & 255] = aff0[row0 * 256 + i];
  __syncthreads();

#pragma unroll
  for (int rr = 0; rr < 2; ++rr) {       // each wave normalizes 2 rows
    const int r = wave * 2 + rr;
    float s = 0.f, s2 = 0.f;
    for (int i = lane; i < 256; i += 32) { float v = xrow[r][i]; s += v; s2 += v * v; }
    s  += __shfl_xor(s, 1);  s  += __shfl_xor(s, 2);  s  += __shfl_xor(s, 4);
    s  += __shfl_xor(s, 8);  s  += __shfl_xor(s, 16);
    s2 += __shfl_xor(s2, 1); s2 += __shfl_xor(s2, 2); s2 += __shfl_xor(s2, 4);
    s2 += __shfl_xor(s2, 8); s2 += __shfl_xor(s2, 16);
    const float mean = s * (1.f / 256.f);
    const float var  = s2 * (1.f / 256.f) - mean * mean;
    const float rstd = rsqrtf(var + 1e-5f);
    for (int i = lane; i < 256; i += 32)
      hbuf[r][i] = (_Float16)(((xrow[r][i] - mean) * rstd) * lng[i] + lnb[i]);
  }
  __syncthreads();

  // GEMM1: out = h @ w1^T ; each wave does 2 column tiles of 16
  for (int t = wave; t < 16; t += 8) {
    const int col0 = t * 16;
    const float* wp = w1 + (size_t)(col0 + m) * 256;
    v8f acc = {};
#pragma unroll 1
    for (int k0 = 0; k0 < 256; k0 += 32) {
      h8 x0 = *(const h8*)(&hbuf[m][k0 + half * 16]);
      h8 x1 = *(const h8*)(&hbuf[m][k0 + half * 16 + 8]);
      h8 y0 = ldg_h8(wp + k0 + half * 16);
      h8 y1 = ldg_h8(wp + k0 + half * 16 + 8);
      acc = wmma16(mk16(x0, x1), mk16(y0, y1), acc);
    }
    const float bias = b1[col0 + m];
#pragma unroll
    for (int j = 0; j < 8; ++j) {
      float xg = acc[j] + bias;
      xg = 0.5f * xg * (1.f + erff(xg * 0.70710678118f));   // exact GELU
      gbuf[j + half * 8][col0 + m] = (_Float16)xg;
    }
  }
  __syncthreads();

  // GEMM2 + residual
  for (int t = wave; t < 16; t += 8) {
    const int col0 = t * 16;
    const float* wp = w2 + (size_t)(col0 + m) * 256;
    v8f acc = {};
#pragma unroll 1
    for (int k0 = 0; k0 < 256; k0 += 32) {
      h8 x0 = *(const h8*)(&gbuf[m][k0 + half * 16]);
      h8 x1 = *(const h8*)(&gbuf[m][k0 + half * 16 + 8]);
      h8 y0 = ldg_h8(wp + k0 + half * 16);
      h8 y1 = ldg_h8(wp + k0 + half * 16 + 8);
      acc = wmma16(mk16(x0, x1), mk16(y0, y1), acc);
    }
    const float bias = b2[col0 + m];
#pragma unroll
    for (int j = 0; j < 8; ++j) {
      const int r = j + half * 8;
      aff1[(row0 + r) * 256 + col0 + m] = xrow[r][col0 + m] + acc[j] + bias;
    }
  }
}

// ---------------------------------------------------------------------------
// Kernel 3: 3x3 patch conv over P x P with 256 input channels -> y_t.
// ---------------------------------------------------------------------------
__global__ __launch_bounds__(256) void k_pconv(
    const float* __restrict__ aff1, const float* __restrict__ pw,
    const float* __restrict__ pb, float* __restrict__ y_t) {
  __shared__ float wl[9 * 256];          // transposed weights [tap][channel]
  const int tid = threadIdx.x;
  const int b   = blockIdx.x;
  for (int i = tid; i < 2304; i += 256) {
    const int t = i / 9, kkk = i - t * 9;
    wl[kkk * 256 + t] = pw[i];
  }
  __syncthreads();
  if (tid < NL) {
    const int py = tid / NP, px = tid - py * NP;
    float acc = pb[0];
#pragma unroll
    for (int dy = -1; dy <= 1; ++dy) {
#pragma unroll
      for (int dx = -1; dx <= 1; ++dx) {
        const int yy = py + dy, xx = px + dx;
        if (yy < 0 || yy >= NP || xx < 0 || xx >= NP) continue;
        const float* rp = aff1 + ((size_t)b * NL + yy * NP + xx) * 256;
        const float* wp = &wl[((dy + 1) * 3 + (dx + 1)) * 256];
        float s = 0.f;
        for (int t = 0; t < 256; ++t) s += wp[t] * rp[t];
        acc += s;
      }
    }
    y_t[b * NL + tid] = acc;
  }
}

// ---------------------------------------------------------------------------
// Kernel 4: spatial synopsis attention. One block per (b,t).
// QK via WMMA (16 synos x 196 keys x 1024), softmax over keys (logits ~0.02,
// no max-shift needed), then the mean-over-synos collapses the value mix to
// a 196-weight combination of v rows (single coalesced pass over v).
// ---------------------------------------------------------------------------
__global__ __launch_bounds__(256) void k_spatial(
    const float* __restrict__ kk, const float* __restrict__ vv,
    const float* __restrict__ syno, float* __restrict__ y_s) {
  __shared__ _Float16 syn[16 * 1024];    // scaled syno_emb f16, 32 KB
  __shared__ float    saff[16 * 208];    // [syno][key] with padded tail, 13 KB
  __shared__ float    colw[208];

  const int tid  = threadIdx.x;
  const int wave = tid >> 5;
  const int lane = tid & 31;
  const int half = lane >> 4;
  const int m    = lane & 15;
  const int b    = blockIdx.x >> 4;
  const int t    = blockIdx.x & 15;

  for (int i = tid; i < 16384; i += 256)
    syn[i] = (_Float16)(syno[i] * 0.03125f);       // 1/sqrt(1024)
  __syncthreads();

  const float* kf = kk + ((size_t)(b * NT + t) * 197 + 1) * (size_t)NDM;
  for (int tile = wave; tile < 13; tile += 8) {    // 13 tiles of 16 keys
    const int l0 = tile * 16;
    int lc = l0 + m; if (lc > 195) lc = 195;       // clamp tail (never read)
    const float* kp = kf + (size_t)lc * NDM;
    v8f acc = {};
#pragma unroll 1
    for (int k0 = 0; k0 < 1024; k0 += 32) {
      h8 x0 = *(const h8*)(&syn[m * 1024 + k0 + half * 16]);
      h8 x1 = *(const h8*)(&syn[m * 1024 + k0 + half * 16 + 8]);
      h8 y0 = ldg_h8(kp + k0 + half * 16);
      h8 y1 = ldg_h8(kp + k0 + half * 16 + 8);
      acc = wmma16(mk16(x0, x1), mk16(y0, y1), acc);
    }
#pragma unroll
    for (int j = 0; j < 8; ++j)
      saff[(j + half * 8) * 208 + l0 + m] = acc[j];
  }
  __syncthreads();

#pragma unroll
  for (int rr = 0; rr < 2; ++rr) {                 // softmax over 196 keys
    const int rq = wave * 2 + rr;
    float s = 0.f;
    for (int i = lane; i < 196; i += 32) {
      const float e = __expf(saff[rq * 208 + i]);  // logits tiny: no max-shift
      saff[rq * 208 + i] = e; s += e;
    }
    s += __shfl_xor(s, 1); s += __shfl_xor(s, 2); s += __shfl_xor(s, 4);
    s += __shfl_xor(s, 8); s += __shfl_xor(s, 16);
    const float inv = __builtin_amdgcn_rcpf(s);
    for (int i = lane; i < 196; i += 32) saff[rq * 208 + i] *= inv;
  }
  __syncthreads();

  if (tid < 196) {                                  // sum over 16 synos
    float s = 0.f;
#pragma unroll
    for (int qq = 0; qq < 16; ++qq) s += saff[qq * 208 + tid];
    colw[tid] = s;
  }
  __syncthreads();

  const float* vf = vv + ((size_t)(b * NT + t) * 197 + 1) * (size_t)NDM;
  const int w4 = tid * 4;
  float a0 = 0.f, a1 = 0.f, a2 = 0.f, a3 = 0.f;
  for (int l2 = 0; l2 < 196; ++l2) {
    if (l2 + 8 < 196) __builtin_prefetch(vf + (size_t)(l2 + 8) * NDM + w4, 0, 1);
    const float cw = colw[l2];
    const float4 vx = *(const float4*)(vf + (size_t)l2 * NDM + w4);
    a0 += cw * vx.x; a1 += cw * vx.y; a2 += cw * vx.z; a3 += cw * vx.w;
  }
  float* dst = y_s + (size_t)b * NDM + w4;
  const float sc = 1.f / 256.f;                     // mean over T*NS
  atomicAdd(dst + 0, a0 * sc); atomicAdd(dst + 1, a1 * sc);
  atomicAdd(dst + 2, a2 * sc); atomicAdd(dst + 3, a3 * sc);
}

__global__ void k_zero(float* __restrict__ p, int n) {
  const int i = blockIdx.x * 256 + threadIdx.x;
  if (i < n) p[i] = 0.f;
}

// ---------------------------------------------------------------------------
extern "C" void kernel_launch(void* const* d_in, const int* in_sizes, int n_in,
                              void* d_out, int out_size, void* d_ws,
                              size_t ws_size, hipStream_t stream) {
  (void)in_sizes; (void)n_in; (void)out_size; (void)ws_size;
  const float* q    = (const float*)d_in[0];
  const float* k    = (const float*)d_in[1];
  const float* v    = (const float*)d_in[2];
  const float* syno = (const float*)d_in[3];
  const float* tw   = (const float*)d_in[4];
  const float* tb   = (const float*)d_in[5];
  const float* lng  = (const float*)d_in[6];
  const float* lnb  = (const float*)d_in[7];
  const float* w1   = (const float*)d_in[8];
  const float* b1   = (const float*)d_in[9];
  const float* w2   = (const float*)d_in[10];
  const float* b2   = (const float*)d_in[11];
  const float* pw   = (const float*)d_in[12];
  const float* pb   = (const float*)d_in[13];

  float* aff0 = (float*)d_ws;                       // [1568][256]
  float* aff1 = aff0 + (size_t)NB * NL * 256;       // [1568][256]
  float* y_t  = (float*)d_out;                      // [8*196]
  float* y_s  = y_t + NB * NL;                      // [8*1024]

  k_zero<<<dim3((NB * NDM + 255) / 256), dim3(256), 0, stream>>>(y_s, NB * NDM);
  k_temporal<<<dim3(NB * NL), dim3(256), 0, stream>>>(q, k, tw, tb, aff0);
  k_mlp<<<dim3(NB * NL / 16), dim3(256), 0, stream>>>(aff0, lng, lnb, w1, b1,
                                                      w2, b2, aff1);
  k_pconv<<<dim3(NB), dim3(256), 0, stream>>>(aff1, pw, pb, y_t);
  k_spatial<<<dim3(NB * NT), dim3(256), 0, stream>>>(k, v, syno, y_s);
}
